// BAF_49117245997138
// MI455X (gfx1250) — matrix operand order repaired and checked by reading, not compiled, not run on hardware
//
#include <hip/hip_runtime.h>
#include <hip/hip_bf16.h>

typedef __attribute__((ext_vector_type(16))) _Float16 v16h;
typedef __attribute__((ext_vector_type(8)))  _Float16 v8h;
typedef __attribute__((ext_vector_type(8)))  float    v8f;

#define NBANDS 5
#define BATCH  512
#define KTOK   64
#define DDIM   200
#define NH     4
#define HDIM   50     // head dim
#define HID    512
#define FIN    64000  // NBANDS*KTOK*DDIM
#define SEQ    320    // NBANDS*KTOK
#define DPAD   224    // DDIM padded to multiple of 32 (K dim)
#define HDPAD  64     // head dim padded to multiple of 32 (K dim)
#define EPROJ  256    // NH*HDPAD
#define NPAD   208    // DDIM padded to multiple of 16 (N dim)
#define KSLICE 16     // split-K factor for router GEMM
#define KCHUNK 4000   // FIN / KSLICE
#define RBM    128    // router block M
#define RBN    128    // router block N
#define RSTEPS (KCHUNK / 32)  // 125 K-steps per slice

#define WMMA_F32_F16(a, b, c) \
  __builtin_amdgcn_wmma_f32_16x16x32_f16(false, (a), false, (b), (short)0, (c), false, false)

// ---------------------------------------------------------------------------
// Fragment loaders (layouts per CDNA5 ISA 7.12.2, 16-bit A 16x32 / B 32x16).
// A: lane holds row M=lane&15; K = kb..kb+7 and kb+16..kb+23, kb=(lane>>4)*8.
// B (given source transposed, src[n][k]): lane holds col N=lane&15,
//    K = kb..kb+7 and kb+16..kb+23, kb=(lane>>4)*8 (same as A).
// Work for both global and LDS source pointers (ds_load_b128 when shared).
// ---------------------------------------------------------------------------
static __device__ __forceinline__ v16h load_frag_a(const _Float16* tile, int ld) {
  const int lane = threadIdx.x & 31;
  const _Float16* p = tile + (lane & 15) * ld + ((lane >> 4) << 3);
  v16h a;
  ((v8h*)&a)[0] = *(const v8h*)(p);
  ((v8h*)&a)[1] = *(const v8h*)(p + 16);
  return a;
}
static __device__ __forceinline__ v16h load_frag_bt(const _Float16* tile, int ld) {
  const int lane = threadIdx.x & 31;
  const _Float16* p = tile + (lane & 15) * ld + ((lane >> 4) << 3);
  v16h b;
  ((v8h*)&b)[0] = *(const v8h*)(p);
  ((v8h*)&b)[1] = *(const v8h*)(p + 16);
  return b;
}

// ---------------------------------------------------------------------------
// Conversions / fills
// ---------------------------------------------------------------------------
__global__ void k_convert_bands(const float* __restrict__ bands,
                                _Float16* __restrict__ kvr,   // [B][FIN]
                                _Float16* __restrict__ kva) { // [B][SEQ][DPAD]
  const long long total = (long long)BATCH * SEQ * DPAD;
  for (long long idx = blockIdx.x * (long long)blockDim.x + threadIdx.x;
       idx < total; idx += (long long)gridDim.x * blockDim.x) {
    int dd = (int)(idx % DPAD);
    long long bs = idx / DPAD;
    int s = (int)(bs % SEQ);
    int b = (int)(bs / SEQ);
    int n = s / KTOK, k = s % KTOK;
    float v = 0.f;
    if (dd < DDIM)
      v = bands[(((long long)n * BATCH + b) * KTOK + k) * DDIM + dd];
    _Float16 hv = (_Float16)v;
    kva[idx] = hv;
    if (dd < DDIM)
      kvr[(long long)b * FIN + s * DDIM + dd] = hv;
  }
}

__global__ void k_convert_w1(const float* __restrict__ w1,
                             _Float16* __restrict__ w1h) {
  const long long total = (long long)HID * FIN;
  for (long long idx = blockIdx.x * (long long)blockDim.x + threadIdx.x;
       idx < total; idx += (long long)gridDim.x * blockDim.x)
    w1h[idx] = (_Float16)w1[idx];
}

__global__ void k_convert_projw(const float* __restrict__ in_proj_w,
                                const float* __restrict__ out_w,
                                _Float16* __restrict__ wqh,
                                _Float16* __restrict__ wkh,
                                _Float16* __restrict__ wvh,
                                _Float16* __restrict__ woh) {
  const int per = NPAD * DPAD;
  const int total = 4 * per;
  for (int idx = blockIdx.x * blockDim.x + threadIdx.x; idx < total;
       idx += gridDim.x * blockDim.x) {
    int mat = idx / per;
    int rem = idx - mat * per;
    int r = rem / DPAD, c = rem % DPAD;
    float v = 0.f;
    if (r < DDIM && c < DDIM) {
      if (mat == 0)      v = in_proj_w[r * DDIM + c];
      else if (mat == 1) v = in_proj_w[(DDIM + r) * DDIM + c];
      else if (mat == 2) v = in_proj_w[(2 * DDIM + r) * DDIM + c];
      else               v = out_w[r * DDIM + c];
    }
    _Float16* dst = (mat == 0) ? wqh : (mat == 1) ? wkh : (mat == 2) ? wvh : woh;
    dst[rem] = (_Float16)v;
  }
}

__global__ void k_fill_zero_h(_Float16* __restrict__ p, long long n) {
  for (long long i = blockIdx.x * (long long)blockDim.x + threadIdx.x; i < n;
       i += (long long)gridDim.x * blockDim.x)
    p[i] = (_Float16)0.f;
}

// ---------------------------------------------------------------------------
// Router GEMM: h_part[ks] = flat16 @ w1h^T  (split-K, WMMA f16->f32).
// LDS-blocked 128x128 per workgroup (8 waves), double-buffered K=32 panels:
// 16 KB of LDS traffic feeds 64 WMMA tiles per step (64 FLOP/byte vs 8 for
// the naive per-wave scheme) -- keeps the loop WMMA-bound, not L2-bound.
// ---------------------------------------------------------------------------
__global__ void __launch_bounds__(256)
k_router_gemm(const _Float16* __restrict__ A,   // [512][FIN]
              const _Float16* __restrict__ W,   // [512][FIN]
              float* __restrict__ hpart) {      // [KSLICE][512][512]
  __shared__ __align__(16) _Float16 As[2][RBM][32];
  __shared__ __align__(16) _Float16 Bs[2][RBN][32];

  const int tid  = threadIdx.x;
  const int wave = tid >> 5, lane = tid & 31;
  const int mblk = (blockIdx.x >> 2) * RBM;   // 4 m-blocks
  const int nblk = (blockIdx.x & 3) * RBN;    // 4 n-blocks
  const int ks   = blockIdx.y;

  // cooperative loader mapping: thread t -> row t/2, 16-half chunk (t&1)*16
  const int frow = tid >> 1;
  const int fcol = (tid & 1) << 4;
  const _Float16* aG = A + (long long)(mblk + frow) * FIN + ks * KCHUNK + fcol;
  const _Float16* bG = W + (long long)(nblk + frow) * FIN + ks * KCHUNK + fcol;

  v8h a0, a1, b0, b1;
  a0 = *(const v8h*)(aG);     a1 = *(const v8h*)(aG + 8);
  b0 = *(const v8h*)(bG);     b1 = *(const v8h*)(bG + 8);
  *(v8h*)&As[0][frow][fcol] = a0;  *(v8h*)&As[0][frow][fcol + 8] = a1;
  *(v8h*)&Bs[0][frow][fcol] = b0;  *(v8h*)&Bs[0][frow][fcol + 8] = b1;
  __syncthreads();

  v8f acc[8];
#pragma unroll
  for (int i = 0; i < 8; ++i) acc[i] = (v8f){};

  for (int s = 0; s < RSTEPS; ++s) {
    const int cur = s & 1;
    if (s + 1 < RSTEPS) {  // prefetch next panel while computing this one
      const _Float16* ag = aG + (s + 1) * 32;
      const _Float16* bg = bG + (s + 1) * 32;
      a0 = *(const v8h*)(ag);  a1 = *(const v8h*)(ag + 8);
      b0 = *(const v8h*)(bg);  b1 = *(const v8h*)(bg + 8);
    }
    const v16h af = load_frag_a(&As[cur][wave * 16][0], 32);
#pragma unroll
    for (int nt = 0; nt < 8; ++nt) {
      const v16h bf = load_frag_bt(&Bs[cur][nt * 16][0], 32);
      acc[nt] = WMMA_F32_F16(af, bf, acc[nt]);
    }
    if (s + 1 < RSTEPS) {
      // safe: buffer (1-cur) was last *read* in step s-1, fenced by that
      // step's barrier; writes here don't race reads of `cur`.
      const int nxt = 1 - cur;
      *(v8h*)&As[nxt][frow][fcol] = a0;  *(v8h*)&As[nxt][frow][fcol + 8] = a1;
      *(v8h*)&Bs[nxt][frow][fcol] = b0;  *(v8h*)&Bs[nxt][frow][fcol + 8] = b1;
    }
    __syncthreads();
  }

  const int rb = (lane >> 4) << 3;
  float* dst = hpart + (long long)ks * BATCH * HID;
#pragma unroll
  for (int nt = 0; nt < 8; ++nt) {
    const int n = nblk + nt * 16 + (lane & 15);
    const int m = mblk + wave * 16 + rb;
#pragma unroll
    for (int r = 0; r < 8; ++r)
      dst[(m + r) * HID + n] = acc[nt][r];
  }
}

__global__ void k_router_reduce(const float* __restrict__ hpart,
                                const float* __restrict__ b1,
                                float* __restrict__ hout) {
  int idx = blockIdx.x * blockDim.x + threadIdx.x;
  if (idx >= BATCH * HID) return;
  float s = 0.f;
#pragma unroll
  for (int p = 0; p < KSLICE; ++p)
    s += hpart[(long long)p * BATCH * HID + idx];
  s += b1[idx % HID];
  hout[idx] = s > 0.f ? s : 0.f;  // ReLU
}

// logits = h @ w2^T + b2; softmax is monotone -> argmax of logits (first max)
__global__ void k_select(const float* __restrict__ h,
                         const float* __restrict__ w2,
                         const float* __restrict__ b2,
                         int* __restrict__ sel) {
  int b = blockIdx.x * blockDim.x + threadIdx.x;
  if (b >= BATCH) return;
  float acc[NBANDS];
#pragma unroll
  for (int p = 0; p < NBANDS; ++p) acc[p] = b2[p];
  const float* hr = h + (long long)b * HID;
  for (int j = 0; j < HID; ++j) {
    float hv = hr[j];
#pragma unroll
    for (int p = 0; p < NBANDS; ++p) acc[p] += hv * w2[p * HID + j];
  }
  int best = 0;
  float bv = acc[0];
#pragma unroll
  for (int p = 1; p < NBANDS; ++p)
    if (acc[p] > bv) { bv = acc[p]; best = p; }
  sel[b] = best;
}

// ---------------------------------------------------------------------------
// K / V / Q projections (WMMA over padded d=224), remapped into per-head
// padded layouts: kproj/qproj col p = (n/50)*64 + n%50; v stored transposed.
// ---------------------------------------------------------------------------
__global__ void k_kproj(const _Float16* __restrict__ kva,   // [B*SEQ][DPAD]
                        const _Float16* __restrict__ wkh,   // [NPAD][DPAD]
                        const float* __restrict__ in_proj_b,
                        _Float16* __restrict__ kp) {        // [B*SEQ][EPROJ]
  const int lane = threadIdx.x & 31;
  const int mt = blockIdx.x, nt = blockIdx.y;
  const _Float16* arow = kva + (long long)mt * 16 * DPAD;
  const _Float16* brow = wkh + (long long)nt * 16 * DPAD;
  v8f c = {};
  for (int k0 = 0; k0 < DPAD; k0 += 32) {
    v16h a = load_frag_a(arow + k0, DPAD);
    v16h b = load_frag_bt(brow + k0, DPAD);
    c = WMMA_F32_F16(a, b, c);
  }
  const int n = nt * 16 + (lane & 15);
  if (n < DDIM) {
    const float bias = in_proj_b[DDIM + n];
    const int p = (n / HDIM) * HDPAD + (n % HDIM);
    const int rb = mt * 16 + ((lane >> 4) << 3);
#pragma unroll
    for (int r = 0; r < 8; ++r)
      kp[(long long)(rb + r) * EPROJ + p] = (_Float16)(c[r] + bias);
  }
}

__global__ void k_vproj(const _Float16* __restrict__ kva,
                        const _Float16* __restrict__ wvh,
                        const float* __restrict__ in_proj_b,
                        _Float16* __restrict__ vT) {        // [B][NH][HDPAD][SEQ]
  const int lane = threadIdx.x & 31;
  const int mt = blockIdx.x, nt = blockIdx.y;
  const _Float16* arow = kva + (long long)mt * 16 * DPAD;
  const _Float16* brow = wvh + (long long)nt * 16 * DPAD;
  v8f c = {};
  for (int k0 = 0; k0 < DPAD; k0 += 32) {
    v16h a = load_frag_a(arow + k0, DPAD);
    v16h b = load_frag_bt(brow + k0, DPAD);
    c = WMMA_F32_F16(a, b, c);
  }
  const int n = nt * 16 + (lane & 15);
  if (n < DDIM) {
    const float bias = in_proj_b[2 * DDIM + n];
    const int hh = n / HDIM, e = n % HDIM;
    const int rb = mt * 16 + ((lane >> 4) << 3);
#pragma unroll
    for (int r = 0; r < 8; ++r) {
      int row = rb + r;
      int b = row / SEQ, s = row % SEQ;
      vT[(((long long)b * NH + hh) * HDPAD + e) * SEQ + s] = (_Float16)(c[r] + bias);
    }
  }
}

__global__ void k_qproj(const _Float16* __restrict__ kva,
                        const _Float16* __restrict__ wqh,
                        const float* __restrict__ in_proj_b,
                        const int* __restrict__ sel,
                        _Float16* __restrict__ qp) {        // [B][KTOK][EPROJ]
  const int lane = threadIdx.x & 31;
  const int bm = blockIdx.x;            // b*4 + itile
  const int b = bm >> 2, it = bm & 3;
  const int nt = blockIdx.y;
  const int srow = sel[b] * KTOK + it * 16;  // gathered Q rows (contiguous)
  const _Float16* arow = kva + ((long long)b * SEQ + srow) * DPAD;
  const _Float16* brow = wqh + (long long)nt * 16 * DPAD;
  v8f c = {};
  for (int k0 = 0; k0 < DPAD; k0 += 32) {
    v16h a = load_frag_a(arow + k0, DPAD);
    v16h bb = load_frag_bt(brow + k0, DPAD);
    c = WMMA_F32_F16(a, bb, c);
  }
  const int n = nt * 16 + (lane & 15);
  if (n < DDIM) {
    const float bias = in_proj_b[n];
    const int p = (n / HDIM) * HDPAD + (n % HDIM);
    const int rb = it * 16 + ((lane >> 4) << 3);
#pragma unroll
    for (int r = 0; r < 8; ++r)
      qp[((long long)b * KTOK + rb + r) * EPROJ + p] = (_Float16)(c[r] + bias);
  }
}

// ---------------------------------------------------------------------------
// Fused scores (q@k^T * scale) + row softmax -> f16 attn probabilities.
// grid: (B, NH, 4 m-strips of 16 rows); block: 4 waves; LDS 16x320 f32.
// ---------------------------------------------------------------------------
__global__ void k_scores_softmax(const _Float16* __restrict__ qp,
                                 const _Float16* __restrict__ kp,
                                 _Float16* __restrict__ attn) { // [B][NH][KTOK][SEQ]
  __shared__ float sc[16][SEQ];
  __shared__ float red1[16][8];
  __shared__ float red2[16][8];
  const int b = blockIdx.x, h = blockIdx.y, mt = blockIdx.z;
  const int wave = threadIdx.x >> 5, lane = threadIdx.x & 31;
  const float scale = 0.14142135623730950488f;  // 1/sqrt(50)
  const _Float16* abase = qp + ((long long)b * KTOK + mt * 16) * EPROJ + h * HDPAD;
  for (int t = 0; t < 5; ++t) {
    const int nt = wave * 5 + t;  // 20 n-tiles of j
    const _Float16* bbase = kp + ((long long)b * SEQ + nt * 16) * EPROJ + h * HDPAD;
    v8f c = {};
    for (int k0 = 0; k0 < HDPAD; k0 += 32) {
      v16h a = load_frag_a(abase + k0, EPROJ);
      v16h bb = load_frag_bt(bbase + k0, EPROJ);
      c = WMMA_F32_F16(a, bb, c);
    }
    const int n = nt * 16 + (lane & 15);
    const int rb = (lane >> 4) << 3;
#pragma unroll
    for (int r = 0; r < 8; ++r) sc[rb + r][n] = c[r] * scale;
  }
  __syncthreads();
  const int row = threadIdx.x >> 3;  // 0..15
  const int t8 = threadIdx.x & 7;
  float mx = -1e30f;
  for (int j = t8; j < SEQ; j += 8) mx = fmaxf(mx, sc[row][j]);
  red1[row][t8] = mx;
  __syncthreads();
#pragma unroll
  for (int q = 0; q < 8; ++q) mx = fmaxf(mx, red1[row][q]);
  float sum = 0.f;
  for (int j = t8; j < SEQ; j += 8) {
    float e = __expf(sc[row][j] - mx);
    sc[row][j] = e;
    sum += e;
  }
  red2[row][t8] = sum;
  __syncthreads();
  sum = 0.f;
#pragma unroll
  for (int q = 0; q < 8; ++q) sum += red2[row][q];
  const float inv = 1.f / sum;
  _Float16* arow = attn + (((long long)b * NH + h) * KTOK + mt * 16 + row) * SEQ;
  for (int j = t8; j < SEQ; j += 8) arow[j] = (_Float16)(sc[row][j] * inv);
}

// ---------------------------------------------------------------------------
// o = attn @ v : M=64(i), N=64(e pad), K=320(j). grid (B,NH), 4 waves = m-tiles.
// ---------------------------------------------------------------------------
__global__ void k_av(const _Float16* __restrict__ attn,
                     const _Float16* __restrict__ vT,
                     _Float16* __restrict__ o16) {          // [B][KTOK][DPAD]
  const int b = blockIdx.x, h = blockIdx.y;
  const int wave = threadIdx.x >> 5, lane = threadIdx.x & 31;
  const int mt = wave;
  const _Float16* abase = attn + (((long long)b * NH + h) * KTOK + mt * 16) * SEQ;
  for (int nt = 0; nt < 4; ++nt) {
    const _Float16* bbase = vT + (((long long)b * NH + h) * HDPAD + nt * 16) * SEQ;
    v8f c = {};
    for (int k0 = 0; k0 < SEQ; k0 += 32) {
      v16h a = load_frag_a(abase + k0, SEQ);
      v16h bb = load_frag_bt(bbase + k0, SEQ);
      c = WMMA_F32_F16(a, bb, c);
    }
    const int e = nt * 16 + (lane & 15);
    if (e < HDIM) {
      const int ib = mt * 16 + ((lane >> 4) << 3);
#pragma unroll
      for (int r = 0; r < 8; ++r)
        o16[((long long)b * KTOK + ib + r) * DPAD + h * HDIM + e] = (_Float16)c[r];
    }
  }
}

// ---------------------------------------------------------------------------
// Output projection: out = o @ out_w^T + out_b  (f32 result)
// ---------------------------------------------------------------------------
__global__ void k_outproj(const _Float16* __restrict__ o16,  // [B*KTOK][DPAD]
                          const _Float16* __restrict__ woh,  // [NPAD][DPAD]
                          const float* __restrict__ out_b,
                          float* __restrict__ out) {         // [B*KTOK][DDIM]
  const int lane = threadIdx.x & 31;
  const int mt = blockIdx.x, nt = blockIdx.y;
  const _Float16* arow = o16 + (long long)mt * 16 * DPAD;
  const _Float16* brow = woh + (long long)nt * 16 * DPAD;
  v8f c = {};
  for (int k0 = 0; k0 < DPAD; k0 += 32) {
    v16h a = load_frag_a(arow + k0, DPAD);
    v16h b = load_frag_bt(brow + k0, DPAD);
    c = WMMA_F32_F16(a, b, c);
  }
  const int n = nt * 16 + (lane & 15);
  if (n < DDIM) {
    const float bias = out_b[n];
    const int rb = mt * 16 + ((lane >> 4) << 3);
#pragma unroll
    for (int r = 0; r < 8; ++r)
      out[(long long)(rb + r) * DDIM + n] = c[r] + bias;
  }
}

// ---------------------------------------------------------------------------
extern "C" void kernel_launch(void* const* d_in, const int* in_sizes, int n_in,
                              void* d_out, int out_size, void* d_ws, size_t ws_size,
                              hipStream_t stream) {
  (void)in_sizes; (void)n_in; (void)out_size; (void)ws_size;
  const float* bands     = (const float*)d_in[0];
  const float* w1        = (const float*)d_in[1];
  const float* b1        = (const float*)d_in[2];
  const float* w2        = (const float*)d_in[3];
  const float* b2        = (const float*)d_in[4];
  const float* in_proj_w = (const float*)d_in[5];
  const float* in_proj_b = (const float*)d_in[6];
  const float* out_w     = (const float*)d_in[7];
  const float* out_b     = (const float*)d_in[8];
  float* out = (float*)d_out;

  char* ws = (char*)d_ws;
  size_t off = 0;
  auto take = [&](size_t bytes) -> char* {
    char* p = ws + off;
    off = (off + bytes + 255) & ~(size_t)255;
    return p;
  };
  _Float16* kvr   = (_Float16*)take((size_t)BATCH * FIN * 2);          // 65.5 MB
  _Float16* kva   = (_Float16*)take((size_t)BATCH * SEQ * DPAD * 2);   // 73.4 MB
  _Float16* w1h   = (_Float16*)take((size_t)HID * FIN * 2);            // 65.5 MB
  float*    hpart = (float*)take((size_t)KSLICE * BATCH * HID * 4);    // 16.8 MB
  float*    hbuf  = (float*)take((size_t)BATCH * HID * 4);             //  1.0 MB
  int*      sel   = (int*)take((size_t)BATCH * 4);
  _Float16* wqh   = (_Float16*)take((size_t)NPAD * DPAD * 2);
  _Float16* wkh   = (_Float16*)take((size_t)NPAD * DPAD * 2);
  _Float16* wvh   = (_Float16*)take((size_t)NPAD * DPAD * 2);
  _Float16* woh   = (_Float16*)take((size_t)NPAD * DPAD * 2);
  _Float16* qp    = (_Float16*)take((size_t)BATCH * KTOK * EPROJ * 2); // 16.8 MB
  _Float16* kp    = (_Float16*)take((size_t)BATCH * SEQ * EPROJ * 2);  // 83.9 MB
  _Float16* vT    = (_Float16*)take((size_t)BATCH * NH * HDPAD * SEQ * 2); // 83.9 MB
  _Float16* attn  = (_Float16*)take((size_t)BATCH * NH * KTOK * SEQ * 2);  // 83.9 MB
  _Float16* o16   = (_Float16*)take((size_t)BATCH * KTOK * DPAD * 2);  // 14.7 MB

  // Stage 0: precision conversion + padding (pads must be zero: K-dims of GEMMs)
  k_convert_bands<<<8192, 256, 0, stream>>>(bands, kvr, kva);
  k_convert_w1<<<8192, 256, 0, stream>>>(w1, w1h);
  k_convert_projw<<<728, 256, 0, stream>>>(in_proj_w, out_w, wqh, wkh, wvh, woh);
  k_fill_zero_h<<<4096, 256, 0, stream>>>(qp, (long long)BATCH * KTOK * EPROJ);
  k_fill_zero_h<<<4096, 256, 0, stream>>>(kp, (long long)BATCH * SEQ * EPROJ);
  k_fill_zero_h<<<4096, 256, 0, stream>>>(o16, (long long)BATCH * KTOK * DPAD);

  // Stage 1: router MLP (LDS-blocked split-K WMMA) -> band selection
  k_router_gemm<<<dim3(16, KSLICE), 256, 0, stream>>>(kvr, w1h, hpart);
  k_router_reduce<<<(BATCH * HID + 255) / 256, 256, 0, stream>>>(hpart, b1, hbuf);
  k_select<<<2, 256, 0, stream>>>(hbuf, w2, b2, sel);

  // Stage 2: Q/K/V projections (WMMA)
  k_kproj<<<dim3(BATCH * SEQ / 16, NPAD / 16), 32, 0, stream>>>(kva, wkh, in_proj_b, kp);
  k_vproj<<<dim3(BATCH * SEQ / 16, NPAD / 16), 32, 0, stream>>>(kva, wvh, in_proj_b, vT);
  k_qproj<<<dim3(BATCH * 4, NPAD / 16), 32, 0, stream>>>(kva, wqh, in_proj_b, sel, qp);

  // Stage 3: attention (WMMA scores + fused softmax, WMMA AV)
  k_scores_softmax<<<dim3(BATCH, NH, 4), 128, 0, stream>>>(qp, kp, attn);
  k_av<<<dim3(BATCH, NH), 128, 0, stream>>>(attn, vT, o16);

  // Stage 4: output projection (WMMA, f32 out)
  k_outproj<<<dim3(BATCH * KTOK / 16, NPAD / 16), 32, 0, stream>>>(o16, woh, out_b, out);
}
